// S6_3315714752559
// MI455X (gfx1250) — matrix operand order repaired and checked by reading, not compiled, not run on hardware
//
#include <hip/hip_runtime.h>
#include <math.h>

typedef __bf16 bf16_t;
typedef bf16_t bf16x4  __attribute__((ext_vector_type(4)));
typedef bf16_t bf16x8  __attribute__((ext_vector_type(8)));
typedef bf16_t bf16x16 __attribute__((ext_vector_type(16)));
typedef float  f32x4   __attribute__((ext_vector_type(4)));
typedef float  f32x8   __attribute__((ext_vector_type(8)));
typedef __attribute__((address_space(3))) bf16_t lds_bf16;

// fragment = two 16B halves occupying 8 contiguous VGPRs; no element shuffles
union frag_u {
    bf16x16 f;
    struct { bf16x8 lo; bf16x8 hi; } h;
};

#define DDIM 1024
#define NSTATE 16
#define NROWS 8192   // b*l = 4*2048

// ---- main-GEMM tiling ----
#define BM 128       // rows per block
#define BN 128       // cols per block
#define BK 64        // K per stage
#define LDP 72       // padded LDS row stride (64 + 8 bf16) -> conflict-free
#define NKB (DDIM / BK)   // 16 K-stages

// ---------------------------------------------------------------- convert f32 -> bf16 (4 elems/thread)
__global__ void s6_cvt_kernel(const float* __restrict__ x,
                              const float* __restrict__ W1,
                              const float* __restrict__ W2,
                              const float* __restrict__ W3,
                              bf16_t* __restrict__ xb,
                              bf16_t* __restrict__ W1b,
                              bf16_t* __restrict__ W2b,
                              bf16_t* __restrict__ W3b) {
    size_t q = (size_t)blockIdx.x * blockDim.x + threadIdx.x;   // quad index
    const size_t nx  = ((size_t)NROWS * DDIM) / 4;
    const size_t nw1 = ((size_t)DDIM * DDIM) / 4;
    const size_t nw  = ((size_t)NSTATE * DDIM) / 4;
    const float* src;
    bf16_t* dst;
    if (q < nx)            { src = x;  dst = xb;  }
    else if ((q -= nx)  < nw1) { src = W1; dst = W1b; }
    else if ((q -= nw1) < nw)  { src = W2; dst = W2b; }
    else if ((q -= nw)  < nw)  { src = W3; dst = W3b; }
    else return;
    f32x4 v = *(const f32x4*)(src + q * 4);
    bf16x4 o;
    o[0] = (bf16_t)v[0]; o[1] = (bf16_t)v[1];
    o[2] = (bf16_t)v[2]; o[3] = (bf16_t)v[3];
    *(bf16x4*)(dst + q * 4) = o;
}

// ---------------------------------------------------------------- A fragment from global (rowscale kernel)
__device__ __forceinline__ bf16x16 load_a_frag_g(const bf16_t* __restrict__ xrow,
                                                 int kk, int half) {
    frag_u a;
    a.h.lo = *(const bf16x8*)(xrow + kk + half * 8);
    a.h.hi = *(const bf16x8*)(xrow + kk + 16 + half * 8);
    return a.f;
}

// ---------------------------------------------------------------- s[r] = dot(B_row + b2, C_row + b3) over n=16
__global__ void s6_rowscale_kernel(const bf16_t* __restrict__ xb,
                                   const bf16_t* __restrict__ W2b,
                                   const bf16_t* __restrict__ W3b,
                                   const float* __restrict__ b2,
                                   const float* __restrict__ b3,
                                   float* __restrict__ s) {
    const int lane = threadIdx.x & 31;
    const int wave = threadIdx.x >> 5;
    const int m    = lane & 15;
    const int half = lane >> 4;
    const int rowBase = (blockIdx.x * 8 + wave) * 16;

    f32x8 accB = {};
    f32x8 accC = {};
    const bf16_t* xrow = xb + (size_t)(rowBase + m) * DDIM;
    const bf16_t* wB   = W2b + (size_t)m * DDIM;
    const bf16_t* wC   = W3b + (size_t)m * DDIM;

    for (int kk = 0; kk < DDIM; kk += 32) {
        bf16x16 a = load_a_frag_g(xrow, kk, half);
        frag_u fB, fC;
        fB.h.lo = *(const bf16x8*)(wB + kk + half * 8);
        fB.h.hi = *(const bf16x8*)(wB + kk + 16 + half * 8);
        fC.h.lo = *(const bf16x8*)(wC + kk + half * 8);
        fC.h.hi = *(const bf16x8*)(wC + kk + 16 + half * 8);
        accB = __builtin_amdgcn_wmma_f32_16x16x32_bf16(false, a, false, fB.f,
                                                       (short)0, accB, false, false);
        accC = __builtin_amdgcn_wmma_f32_16x16x32_bf16(false, a, false, fC.f,
                                                       (short)0, accC, false, false);
    }

    const float bias2 = b2[m];
    const float bias3 = b3[m];
#pragma unroll
    for (int j = 0; j < 8; ++j) {
        float p = (accB[j] + bias2) * (accC[j] + bias3);
        p += __shfl_xor(p, 1);
        p += __shfl_xor(p, 2);
        p += __shfl_xor(p, 4);
        p += __shfl_xor(p, 8);
        if (m == 0) s[rowBase + j + half * 8] = p;
    }
}

// ---------------------------------------------------------------- fast stable softplus (v_exp_f32 / v_log_f32)
__device__ __forceinline__ float softplus_f(float z) {
    // max(z,0) + log1p(exp(-|z|)); t in (0,1] so log(1+t) is well-conditioned
    float t = __expf(-fabsf(z));
    return fmaxf(z, 0.0f) + __logf(1.0f + t);
}

// ---------------------------------------------------------------- async stage: 128x64 bf16 tile -> LDS
__device__ __forceinline__ void s6_stage_async(const bf16_t* __restrict__ gbase,
                                               bf16_t* lbase, int tid) {
#pragma unroll
    for (int i = 0; i < 4; ++i) {
        const int c   = tid + i * 256;      // 0..1023 chunk id
        const int row = c >> 3;             // 8 x 16B chunks per 128B row
        const int seg = c & 7;
        const bf16_t* g = gbase + (size_t)row * DDIM + seg * 8;
        lds_bf16* l = (lds_bf16*)(lbase + row * LDP + seg * 8);
        asm volatile("global_load_async_to_lds_b128 %0, %1, off"
                     :: "v"(l), "v"(g) : "memory");
    }
}

// ---------------------------------------------------------------- main: y = x * softplus(x W1^T + b1) * s
__global__ void __launch_bounds__(256)
s6_main_kernel(const bf16_t* __restrict__ xb,
               const bf16_t* __restrict__ W1b,
               const float* __restrict__ xf,
               const float* __restrict__ b1,
               const float* __restrict__ s,
               float* __restrict__ y) {
    __shared__ bf16_t sA[2][BM][LDP];   // x tile:  rows x K
    __shared__ bf16_t sB[2][BN][LDP];   // W1 tile: cols x K

    const int tid  = threadIdx.x;
    const int lane = tid & 31;
    const int wave = tid >> 5;
    const int m    = lane & 15;
    const int half = lane >> 4;

    // 8 waves: 4 row groups x 2 col groups; wave tile = 32 rows x 64 cols
    const int wr = wave >> 1;             // 0..3
    const int wc = wave & 1;              // 0..1
    const int rowBlock = blockIdx.y * BM;
    const int colBlock = blockIdx.x * BN;

    f32x8 acc[2][4];
#pragma unroll
    for (int ar = 0; ar < 2; ++ar)
#pragma unroll
        for (int ac = 0; ac < 4; ++ac) acc[ar][ac] = (f32x8){};

    // prologue: stage K-block 0 into buffer 0
    s6_stage_async(xb  + (size_t)rowBlock * DDIM, &sA[0][0][0], tid);
    s6_stage_async(W1b + (size_t)colBlock * DDIM, &sB[0][0][0], tid);

    for (int kb = 0; kb < NKB; ++kb) {
        const int cur = kb & 1;
        if (kb + 1 < NKB) {
            const int nxt = cur ^ 1;
            const int k0  = (kb + 1) * BK;
            s6_stage_async(xb  + (size_t)rowBlock * DDIM + k0, &sA[nxt][0][0], tid);
            s6_stage_async(W1b + (size_t)colBlock * DDIM + k0, &sB[nxt][0][0], tid);
            // 8 per-thread loads outstanding for nxt; retire the 8 for cur
            asm volatile("s_wait_asynccnt 0x8" ::: "memory");
        } else {
            asm volatile("s_wait_asynccnt 0x0" ::: "memory");
        }
        __syncthreads();    // cur buffer fully in LDS for all waves

#pragma unroll
        for (int kk = 0; kk < BK; kk += 32) {
            // A fragments (2 row sub-tiles of 16)
            frag_u afr[2];
#pragma unroll
            for (int ar = 0; ar < 2; ++ar) {
                const int r = wr * 32 + ar * 16 + m;
                afr[ar].h.lo = *(const bf16x8*)&sA[cur][r][kk + half * 8];
                afr[ar].h.hi = *(const bf16x8*)&sA[cur][r][kk + 16 + half * 8];
            }
            // B fragments (4 col sub-tiles of 16) + 8 WMMAs
#pragma unroll
            for (int ac = 0; ac < 4; ++ac) {
                const int cidx = wc * 64 + ac * 16 + m;
                frag_u bf;
                bf.h.lo = *(const bf16x8*)&sB[cur][cidx][kk + half * 8];
                bf.h.hi = *(const bf16x8*)&sB[cur][cidx][kk + 16 + half * 8];
#pragma unroll
                for (int ar = 0; ar < 2; ++ar) {
                    acc[ar][ac] = __builtin_amdgcn_wmma_f32_16x16x32_bf16(
                        false, afr[ar].f, false, bf.f, (short)0, acc[ar][ac], false, false);
                }
            }
        }
        __syncthreads();    // all waves done reading cur before it is overwritten
    }

    // epilogue: y[r,e] = x[r,e] * softplus(L + b1[e]) * s[r]
#pragma unroll
    for (int ac = 0; ac < 4; ++ac) {
        const int col  = colBlock + wc * 64 + ac * 16 + m;
        const float bc = b1[col];
#pragma unroll
        for (int ar = 0; ar < 2; ++ar) {
#pragma unroll
            for (int j = 0; j < 8; ++j) {
                const int row  = rowBlock + wr * 32 + ar * 16 + j + half * 8;
                const size_t o = (size_t)row * DDIM + col;
                y[o] = xf[o] * softplus_f(acc[ar][ac][j] + bc) * s[row];
            }
        }
    }
}

// ---------------------------------------------------------------- launch
extern "C" void kernel_launch(void* const* d_in, const int* in_sizes, int n_in,
                              void* d_out, int out_size, void* d_ws, size_t ws_size,
                              hipStream_t stream) {
    const float* x  = (const float*)d_in[0];
    const float* W1 = (const float*)d_in[1];
    const float* b1 = (const float*)d_in[2];
    const float* W2 = (const float*)d_in[3];
    const float* b2 = (const float*)d_in[4];
    const float* W3 = (const float*)d_in[5];
    const float* b3 = (const float*)d_in[6];
    // d_in[7] = A is mathematically dead in the reference (h starts at zero).
    float* y = (float*)d_out;

    char* ws = (char*)d_ws;
    bf16_t* xb  = (bf16_t*)(ws);
    bf16_t* W1b = (bf16_t*)(ws + (size_t)16777216);
    bf16_t* W2b = (bf16_t*)(ws + (size_t)16777216 + 2097152);
    bf16_t* W3b = (bf16_t*)(ws + (size_t)16777216 + 2097152 + 32768);
    float*  sC  = (float*) (ws + (size_t)16777216 + 2097152 + 65536);

    {
        const size_t totalQuads = ((size_t)NROWS * DDIM + (size_t)DDIM * DDIM +
                                   2ull * NSTATE * DDIM) / 4;   // 2,367,488
        const int blocks = (int)((totalQuads + 255) / 256);     // 9,248
        s6_cvt_kernel<<<blocks, 256, 0, stream>>>(x, W1, W2, W3, xb, W1b, W2b, W3b);
    }
    s6_rowscale_kernel<<<64, 256, 0, stream>>>(xb, W2b, W3b, b2, b3, sC);
    s6_main_kernel<<<dim3(DDIM / BN, NROWS / BM), 256, 0, stream>>>(xb, W1b, x, b1, sC, y);
}
